// OnsagerNet_2920577761396
// MI455X (gfx1250) — compile-verified, matrix operand order checked by reading, not run on hardware
//
#include <hip/hip_runtime.h>
#include <hip/hip_bf16.h>

typedef __bf16 bf16;
typedef __bf16 v16bf __attribute__((ext_vector_type(16)));
typedef float  v8f   __attribute__((ext_vector_type(8)));
typedef unsigned int u32x4 __attribute__((ext_vector_type(4)));
typedef int i32x8 __attribute__((ext_vector_type(8)));
typedef int i32x4 __attribute__((ext_vector_type(4)));

union ABfrag { v16bf v; uint4 u4[2]; };

#if __has_builtin(__builtin_amdgcn_global_load_async_to_lds_b128)
#define HAVE_ASYNC_LDS 1
#else
#define HAVE_ASYNC_LDS 0
#endif
#if __has_builtin(__builtin_amdgcn_tensor_load_to_lds)
#define HAVE_TDM 1
#else
#define HAVE_TDM 0
#endif

// async-copy builtin pointer types: global source = AS1 int4*, LDS dest = AS3 int4*
typedef __attribute__((address_space(1))) i32x4* gv4ptr;
typedef __attribute__((address_space(3))) i32x4* lv4ptr;
static __device__ __forceinline__ gv4ptr gvcast(const void* p) {
    return (gv4ptr)(unsigned long long)(uintptr_t)p;
}
static __device__ __forceinline__ lv4ptr lvcast(const void* p) {
    // generic LDS address: low 32 bits are the wave-relative LDS byte offset
    return (lv4ptr)(unsigned)(uintptr_t)p;
}

static __device__ __forceinline__ void wait_async0() {
#if __has_builtin(__builtin_amdgcn_s_wait_asynccnt)
    __builtin_amdgcn_s_wait_asynccnt((short)0);
#else
    asm volatile("s_wait_asynccnt 0" ::: "memory");
#endif
}
static __device__ __forceinline__ void wait_tensor0() {
#if __has_builtin(__builtin_amdgcn_s_wait_tensorcnt)
    __builtin_amdgcn_s_wait_tensorcnt((short)0);
#else
    asm volatile("s_wait_tensorcnt 0" ::: "memory");
#endif
}

static __device__ inline v8f zero8() {
    v8f z;
    #pragma unroll
    for (int i = 0; i < 8; ++i) z[i] = 0.f;
    return z;
}
static __device__ inline float f_act(float z) {
    float r  = fmaxf(z, 0.f);
    float r2 = fmaxf(z - 0.5f, 0.f);
    return r * r - r2 * r2;
}
static __device__ inline float f_actp(float z) {
    return 2.f * fminf(fmaxf(z, 0.f), 0.5f);   // 2*clamp(z,0,0.5)
}

// ---------------------------------------------------------------------------
// small helper kernels
// ---------------------------------------------------------------------------
__global__ void cvt_f32_bf16(const float* __restrict__ src, bf16* __restrict__ dst, int n) {
    int i = blockIdx.x * blockDim.x + threadIdx.x;
    if (i < n) dst[i] = (bf16)src[i];
}
__global__ void cvt_transpose(const float* __restrict__ src, int R, int C, bf16* __restrict__ dst) {
    int i = blockIdx.x * blockDim.x + threadIdx.x;
    if (i < R * C) {
        int r = i / C, c = i % C;
        dst[(size_t)c * R + r] = (bf16)src[i];
    }
}
__global__ void build_xcat(const float* __restrict__ x, const float* __restrict__ T,
                           bf16* __restrict__ xc, int n) {
    int i = blockIdx.x * blockDim.x + threadIdx.x;
    if (i < n) {
        int r = i >> 7, c = i & 127;
        float v = (c < 64) ? x[(size_t)r * 64 + c] : T[(size_t)r * 64 + (c - 64)];
        xc[i] = (bf16)v;
    }
}

// ---------------------------------------------------------------------------
// Generic bf16 WMMA GEMM:  C[M,N] = A[M,K] * Wform[N,K]^T  (+epilogue)
// Block 256 thr = 8 waves; tile 128(M) x 64(N); K-step 32; double-buffered LDS
// filled by GLOBAL_LOAD_ASYNC_TO_LDS (A) and TDM TENSOR_LOAD_TO_LDS (B).
// ---------------------------------------------------------------------------
__global__ __launch_bounds__(256) void gemm_bf16_wmma(
    const bf16* __restrict__ A, int lda,
    const bf16* __restrict__ W, int ldw,
    const float* __restrict__ bias,
    int M, int N, int K, int mode,
    float* __restrict__ outF,
    bf16* __restrict__ outH,
    bf16* __restrict__ outD,
    const bf16* __restrict__ skip,
    const float* __restrict__ dprev,
    const bf16* __restrict__ dact,
    const float* __restrict__ addf,
    float fscale)
{
    __shared__ bf16 As[2][128][48];   // 32 K + 16 pad -> 96B row stride
    __shared__ bf16 Bs[2][64][48];

    const int tid   = threadIdx.x;
    const int wid   = tid >> 5;
    const int lane  = tid & 31;
    const int lr    = lane & 15;
    const int kh    = lane >> 4;
    const int tileN = blockIdx.x * 64;
    const int tileM = blockIdx.y * 128;

    const int ra = tid >> 1, ca = (tid & 1) * 16;
    const bf16* gA = A + (size_t)(tileM + ra) * lda + ca;
    const bf16* gB = W + (size_t)(tileN + ra) * ldw + ca;   // used when tid<128

    __builtin_prefetch(W + (size_t)tileN * ldw, 0, 1);
    __builtin_prefetch(A + (size_t)tileM * lda, 0, 1);

    v8f acc[4];
    #pragma unroll
    for (int n0 = 0; n0 < 4; ++n0) acc[n0] = zero8();

#if HAVE_ASYNC_LDS
    auto stage = [&](int buf, int k0) {
        // A tile 128x32: every thread copies 32B straight into LDS (ASYNCcnt)
        __builtin_amdgcn_global_load_async_to_lds_b128(
            gvcast(gA + k0),     lvcast(&As[buf][ra][ca]), 0, 0);
        __builtin_amdgcn_global_load_async_to_lds_b128(
            gvcast(gA + k0 + 8), lvcast(&As[buf][ra][ca + 8]), 0, 0);
#if HAVE_TDM
        // B tile 64x32 via Tensor Data Mover, one wave issues the DMA.
        // pad_enable: 16 DWORDs data then 8 DWORDs pad -> 96B LDS row stride.
        if (wid == 0) {
            unsigned long long ga = (unsigned long long)(uintptr_t)(W + (size_t)tileN * ldw + k0);
            unsigned ldsb = (unsigned)(uintptr_t)&Bs[buf][0][0];
            u32x4 g0 = {0u, 0u, 0u, 0u};
            i32x8 g1 = {0, 0, 0, 0, 0, 0, 0, 0};
            i32x4 g2 = {0, 0, 0, 0}, g3 = {0, 0, 0, 0};
            i32x8 g4 = {0, 0, 0, 0, 0, 0, 0, 0};
            g0[0] = 1u;                                    // count = 1 (valid)
            g0[1] = ldsb;                                  // lds_addr
            g0[2] = (unsigned)ga;                          // global_addr[31:0]
            g0[3] = (unsigned)((ga >> 32) & 0x01FFFFFFull) // global_addr[56:32]
                    | (2u << 30);                          // type = 2 (image)
            unsigned uk = (unsigned)K, un = (unsigned)N;
            g1[0] = (int)((1u << 16)      // data_size = 1 (2 bytes)
                    | (1u << 20)          // pad_enable
                    | (3u << 22)          // pad_interval: 16 DWORDs
                    | (7u << 25));        // pad_amount:   8 DWORDs
            g1[1] = (int)((uk & 0xFFFFu) << 16);                            // tensor_dim0 lo
            g1[2] = (int)(((uk >> 16) & 0xFFFFu) | ((un & 0xFFFFu) << 16)); // dim0 hi | dim1 lo
            g1[3] = (int)(((un >> 16) & 0xFFFFu) | (32u << 16));            // dim1 hi | tile_dim0=32
            g1[4] = (int)64u;                                               // tile_dim1 = 64
            g1[5] = (int)(unsigned)ldw;                                     // tensor_dim0_stride
            __builtin_amdgcn_tensor_load_to_lds(g0, g1, g2, g3, g4, 0);
        }
#else
        if (tid < 128) {
            __builtin_amdgcn_global_load_async_to_lds_b128(
                gvcast(gB + k0),     lvcast(&Bs[buf][ra][ca]), 0, 0);
            __builtin_amdgcn_global_load_async_to_lds_b128(
                gvcast(gB + k0 + 8), lvcast(&Bs[buf][ra][ca + 8]), 0, 0);
        }
#endif
    };

    stage(0, 0);
    for (int k0 = 0; k0 < K; k0 += 32) {
        const int buf = (k0 >> 5) & 1;
        wait_async0();
#if HAVE_TDM
        if (wid == 0) wait_tensor0();
#endif
        __syncthreads();
        if (k0 + 32 < K) stage(buf ^ 1, k0 + 32);

        ABfrag a;
        a.u4[0] = *(const uint4*)&As[buf][wid * 16 + lr][kh * 16];
        a.u4[1] = *(const uint4*)&As[buf][wid * 16 + lr][kh * 16 + 8];
        #pragma unroll
        for (int n0 = 0; n0 < 4; ++n0) {
            ABfrag b;
            b.u4[0] = *(const uint4*)&Bs[buf][n0 * 16 + lr][kh * 16];
            b.u4[1] = *(const uint4*)&Bs[buf][n0 * 16 + lr][kh * 16 + 8];
            acc[n0] = __builtin_amdgcn_wmma_f32_16x16x32_bf16(
                false, a.v, false, b.v, (short)0, acc[n0], false, false);
        }
    }
#else
    // fallback: synchronous staging
    for (int k0 = 0; k0 < K; k0 += 32) {
        __syncthreads();
        {
            const uint4* gp = (const uint4*)(gA + k0);
            *(uint4*)&As[0][ra][ca]     = gp[0];
            *(uint4*)&As[0][ra][ca + 8] = gp[1];
        }
        if (tid < 128) {
            const uint4* gp = (const uint4*)(gB + k0);
            *(uint4*)&Bs[0][ra][ca]     = gp[0];
            *(uint4*)&Bs[0][ra][ca + 8] = gp[1];
        }
        __syncthreads();
        ABfrag a;
        a.u4[0] = *(const uint4*)&As[0][wid * 16 + lr][kh * 16];
        a.u4[1] = *(const uint4*)&As[0][wid * 16 + lr][kh * 16 + 8];
        #pragma unroll
        for (int n0 = 0; n0 < 4; ++n0) {
            ABfrag b;
            b.u4[0] = *(const uint4*)&Bs[0][n0 * 16 + lr][kh * 16];
            b.u4[1] = *(const uint4*)&Bs[0][n0 * 16 + lr][kh * 16 + 8];
            acc[n0] = __builtin_amdgcn_wmma_f32_16x16x32_bf16(
                false, a.v, false, b.v, (short)0, acc[n0], false, false);
        }
    }
#endif

    // epilogue: lane holds col = lr, rows kh*8 + r  (16x16 f32 C/D layout)
    #pragma unroll
    for (int n0 = 0; n0 < 4; ++n0) {
        #pragma unroll
        for (int r = 0; r < 8; ++r) {
            int row = tileM + wid * 16 + kh * 8 + r;
            int col = tileN + n0 * 16 + lr;
            float z = acc[n0][r] + (bias ? bias[col] : 0.f);
            size_t o = (size_t)row * N + col;
            if (mode == 0) {
                outF[o] = z;
            } else if (mode == 1) {
                float h = f_act(z);
                if (skip) h += (float)skip[o];
                outH[o] = (bf16)h;
                outD[o] = (bf16)f_actp(z);
            } else if (mode == 2) {
                float dh = z + (dprev ? dprev[o] : 0.f);
                outF[o] = dh;
                outH[o] = (bf16)(dh * (float)dact[o]);
            } else {
                outF[o] = fscale * (z + (addf ? addf[o] : 0.f));
            }
        }
    }
}

// ---------------------------------------------------------------------------
// potential fixup (64x64 wpotlin terms), one block (64 thr) per sample
// ---------------------------------------------------------------------------
__global__ __launch_bounds__(64) void potfix_kernel(
    const float* __restrict__ potz, const float* __restrict__ x,
    const float* __restrict__ wpotlin, const float* __restrict__ bpotlin,
    bf16* __restrict__ u, float* __restrict__ gxl)
{
    __shared__ float xs[64], ps[64];
    const int s = blockIdx.x, i = threadIdx.x;
    xs[i] = x[(size_t)s * 64 + i];
    __syncthreads();
    float p = potz[(size_t)s * 64 + i] + bpotlin[i];
    #pragma unroll 4
    for (int j = 0; j < 64; ++j) p += wpotlin[i * 64 + j] * xs[j];
    ps[i] = p;
    u[(size_t)s * 64 + i] = (bf16)(2.f * p);
    __syncthreads();
    float gl = 0.2f * xs[i];   // 2 * POT_BETA * x
    #pragma unroll 4
    for (int j = 0; j < 64; ++j) gl += 2.f * ps[j] * wpotlin[j * 64 + i];
    gxl[(size_t)s * 64 + i] = gl;
}

// ---------------------------------------------------------------------------
// Fused final kernel: 16 samples/block; matA kept bf16 in LDS (never in HBM)
//   out_i = AL@(AL^T g) + AU g - AU^T g + 0.1 g + forc
// ---------------------------------------------------------------------------
__global__ __launch_bounds__(256) void onsager_final_kernel(
    const bf16* __restrict__ h2, const bf16* __restrict__ wmatc,
    const float* __restrict__ bmat, const float* __restrict__ g,
    const float* __restrict__ forc, float* __restrict__ out)
{
    __shared__ bf16  Asm[16][4096];
    __shared__ float gsh[16][64];
    __shared__ float s1[16][64];

    const int s0  = blockIdx.x * 16;
    const int tid = threadIdx.x;

    for (int i = tid; i < 16 * 64; i += 256)
        gsh[i >> 6][i & 63] = g[(size_t)(s0 + (i >> 6)) * 64 + (i & 63)];

    const int wid = tid >> 5, lane = tid & 31;
    const int lr = lane & 15, kh = lane >> 4;

    for (int nt = 0; nt < 32; ++nt) {
        int nbase = wid * 512 + nt * 16;
        v8f acc = zero8();
        for (int k0 = 0; k0 < 1024; k0 += 32) {
            ABfrag a, b;
            const uint4* ap = (const uint4*)&h2[(size_t)(s0 + lr) * 1024 + k0 + kh * 16];
            a.u4[0] = ap[0]; a.u4[1] = ap[1];
            const uint4* bp = (const uint4*)&wmatc[(size_t)(nbase + lr) * 1024 + k0 + kh * 16];
            b.u4[0] = bp[0]; b.u4[1] = bp[1];
            acc = __builtin_amdgcn_wmma_f32_16x16x32_bf16(
                false, a.v, false, b.v, (short)0, acc, false, false);
        }
        int col = nbase + lr;
        #pragma unroll
        for (int r = 0; r < 8; ++r)
            Asm[kh * 8 + r][col] = (bf16)(acc[r] + bmat[col]);
    }
    __syncthreads();

    const int s = tid >> 4, l = tid & 15;
    #pragma unroll
    for (int jj = 0; jj < 4; ++jj) {      // s1[j] = sum_{i>=j} A(i,j) g_i
        int j = l * 4 + jj;
        float a = 0.f;
        for (int i = j; i < 64; ++i) a += (float)Asm[s][i * 64 + j] * gsh[s][i];
        s1[s][j] = a;
    }
    __syncthreads();
    #pragma unroll
    for (int ii = 0; ii < 4; ++ii) {
        int i = l * 4 + ii;
        float a = 0.1f * gsh[s][i] + forc[(size_t)(s0 + s) * 64 + i];
        for (int j = 0; j <= i; ++j) a += (float)Asm[s][i * 64 + j] * s1[s][j];
        for (int j = i + 1; j < 64; ++j) a += (float)Asm[s][i * 64 + j] * gsh[s][j];
        for (int j = 0; j < i; ++j) a -= (float)Asm[s][j * 64 + i] * gsh[s][j];
        out[(size_t)(s0 + s) * 64 + i] = a;
    }
}

// ---------------------------------------------------------------------------
// host launcher
// ---------------------------------------------------------------------------
extern "C" void kernel_launch(void* const* d_in, const int* in_sizes, int n_in,
                              void* d_out, int out_size, void* d_ws, size_t ws_size,
                              hipStream_t stream) {
    (void)in_sizes; (void)n_in; (void)out_size; (void)ws_size;

    const size_t B = 16384, H = 1024, NV = 64, IN0 = 128;

    const float* x       = (const float*)d_in[0];
    const float* T       = (const float*)d_in[1];
    const float* w0      = (const float*)d_in[2];
    const float* b0      = (const float*)d_in[3];
    const float* w1      = (const float*)d_in[4];
    const float* b1      = (const float*)d_in[5];
    const float* w2      = (const float*)d_in[6];
    const float* b2      = (const float*)d_in[7];
    const float* wmat    = (const float*)d_in[8];
    const float* bmat    = (const float*)d_in[9];
    const float* wpot    = (const float*)d_in[10];
    const float* bpot    = (const float*)d_in[11];
    const float* wpotlin = (const float*)d_in[12];
    const float* bpotlin = (const float*)d_in[13];
    const float* wf      = (const float*)d_in[14];
    const float* bf      = (const float*)d_in[15];
    float* out = (float*)d_out;

    size_t off = 0;
    char* wsb = (char*)d_ws;
    auto take = [&](size_t elems, size_t esz) -> void* {
        off = (off + 255) & ~(size_t)255;
        void* p = wsb + off;
        off += elems * esz;
        return p;
    };
    bf16* xcat  = (bf16*)take(B * IN0, 2);
    bf16* w0c   = (bf16*)take(H * IN0, 2);
    bf16* w0T   = (bf16*)take(IN0 * H, 2);
    bf16* w1c   = (bf16*)take(H * H, 2);
    bf16* w2c   = (bf16*)take(H * H, 2);
    bf16* w1T   = (bf16*)take(H * H, 2);
    bf16* w2T   = (bf16*)take(H * H, 2);
    bf16* wmatc = (bf16*)take((size_t)NV * NV * H, 2);
    bf16* wpotc = (bf16*)take(NV * H, 2);
    bf16* wpotT = (bf16*)take(H * NV, 2);
    bf16* wfc   = (bf16*)take(NV * H, 2);
    bf16* h0    = (bf16*)take(B * H, 2);
    bf16* h1    = (bf16*)take(B * H, 2);
    bf16* h2    = (bf16*)take(B * H, 2);
    bf16* dt0   = (bf16*)take(B * H, 2);
    bf16* dt1   = (bf16*)take(B * H, 2);
    bf16* dt2   = (bf16*)take(B * H, 2);
    bf16* uu    = (bf16*)take(B * NV, 2);
    bf16* eA    = (bf16*)take(B * H, 2);
    bf16* eB    = (bf16*)take(B * H, 2);
    float* potz = (float*)take(B * NV, 4);
    float* dh   = (float*)take(B * H, 4);
    float* gxl  = (float*)take(B * NV, 4);
    float* gvec = (float*)take(B * NV, 4);
    float* forc = (float*)take(B * NV, 4);

    auto cvt = [&](const float* s, bf16* d, size_t n) {
        cvt_f32_bf16<<<dim3((n + 255) / 256), dim3(256), 0, stream>>>(s, d, (int)n);
    };
    auto cvtT = [&](const float* s, int R, int C, bf16* d) {
        cvt_transpose<<<dim3(((size_t)R * C + 255) / 256), dim3(256), 0, stream>>>(s, R, C, d);
    };

    cvt(w0, w0c, H * IN0);
    cvt(w1, w1c, H * H);
    cvt(w2, w2c, H * H);
    cvt(wmat, wmatc, (size_t)NV * NV * H);
    cvt(wpot, wpotc, NV * H);
    cvt(wf, wfc, NV * H);
    cvtT(w0, (int)H, (int)IN0, w0T);
    cvtT(w1, (int)H, (int)H, w1T);
    cvtT(w2, (int)H, (int)H, w2T);
    cvtT(wpot, (int)NV, (int)H, wpotT);
    build_xcat<<<dim3((B * IN0 + 255) / 256), dim3(256), 0, stream>>>(x, T, xcat, (int)(B * IN0));

    auto gemm = [&](const bf16* A, int lda, const bf16* W, int ldw, const float* bias,
                    int N, int K, int mode, float* oF, bf16* oH, bf16* oD,
                    const bf16* skip, const float* dprev, const bf16* dact,
                    const float* addf, float fscale) {
        dim3 grid(N / 64, (unsigned)(B / 128));
        gemm_bf16_wmma<<<grid, dim3(256), 0, stream>>>(
            A, lda, W, ldw, bias, (int)B, N, K, mode,
            oF, oH, oD, skip, dprev, dact, addf, fscale);
    };

    // forward MLP
    gemm(xcat, 128, w0c, 128, b0, 1024, 128, 1, nullptr, h0, dt0, nullptr, nullptr, nullptr, nullptr, 1.f);
    gemm(h0, 1024, w1c, 1024, b1, 1024, 1024, 1, nullptr, h1, dt1, h0, nullptr, nullptr, nullptr, 1.f);
    gemm(h1, 1024, w2c, 1024, b2, 1024, 1024, 1, nullptr, h2, dt2, h1, nullptr, nullptr, nullptr, 1.f);

    // potential head + fixup
    gemm(h2, 1024, wpotc, 1024, bpot, 64, 1024, 0, potz, nullptr, nullptr, nullptr, nullptr, nullptr, nullptr, 1.f);
    potfix_kernel<<<dim3((unsigned)B), dim3(64), 0, stream>>>(potz, x, wpotlin, bpotlin, uu, gxl);

    // backward chain
    gemm(uu, 64, wpotT, 64, nullptr, 1024, 64, 2, dh, eA, nullptr, nullptr, nullptr, dt2, nullptr, 1.f);
    gemm(eA, 1024, w2T, 1024, nullptr, 1024, 1024, 2, dh, eB, nullptr, nullptr, dh, dt1, nullptr, 1.f);
    gemm(eB, 1024, w1T, 1024, nullptr, 1024, 1024, 2, dh, eA, nullptr, nullptr, dh, dt0, nullptr, 1.f);
    gemm(eA, 1024, w0T, 1024, nullptr, 64, 1024, 3, gvec, nullptr, nullptr, nullptr, nullptr, nullptr, gxl, -1.f);

    // forcing term
    gemm(h2, 1024, wfc, 1024, bf, 64, 1024, 0, forc, nullptr, nullptr, nullptr, nullptr, nullptr, nullptr, 1.f);

    // fused matA + Onsager apply
    onsager_final_kernel<<<dim3((unsigned)(B / 16)), dim3(256), 0, stream>>>(
        h2, wmatc, bmat, gvec, forc, out);
}
